// GraphAwareCuboidAttention_59064390255206
// MI455X (gfx1250) — matrix operand order repaired and checked by reading, not compiled, not run on hardware
//
#include <hip/hip_runtime.h>
#include <hip/hip_bf16.h>

typedef __attribute__((ext_vector_type(16))) __bf16 v16bf;
typedef __attribute__((ext_vector_type(8)))  float  v8f;

#define NTOK 49152
#define PP   256
#define SS   256
#define HH   256
#define DD   32
#define CORE_ROWS 192
#define PX_STR 264   // padded row stride for sPX  (528 B -> conflict-free, 16B aligned)
#define AT_STR 264   // padded row stride for sAtt
#define VT_STRIDE 264 // padded stride for transposed V

union BF16x16 { unsigned short u[16]; uint4 q[2]; v16bf v; };

// native f32 -> bf16 (RNE) via hardware cvt (lowers to v_cvt_pk_bf16_f32 in pairs)
static __device__ __forceinline__ unsigned short f2bf(float f) {
    union { __bf16 b; unsigned short s; } u;
    u.b = (__bf16)f;
    return u.s;
}

static __device__ __forceinline__ v8f wmma_bf16(v16bf a, v16bf b, v8f c) {
    return __builtin_amdgcn_wmma_f32_16x16x32_bf16(false, a, false, b, (short)0, c, false, false);
}

// A-fragment (16x32 bf16) from row-major bf16 LDS: lane holds row (lane&15),
// K pattern = two contiguous 8-element runs at col0 + 8*half and col0 + 16 + 8*half.
static __device__ __forceinline__ v16bf ldsA(const unsigned short* p, int row0, int stride,
                                             int col0, int lane) {
    const int hf  = lane >> 4;
    const int row = row0 + (lane & 15);
    const unsigned short* r = p + row * stride + col0 + hf * 8;
    BF16x16 t;
    t.q[0] = *(const uint4*)(r);
    t.q[1] = *(const uint4*)(r + 16);
    return t.v;
}

// B-fragment (32x16 bf16): B[k][n] = p[(row0+n)*stride + col0 + k]; lane n=lane&15,
// K pattern = two contiguous 8-element runs at 8*half and 16 + 8*half (2x b128).
static __device__ __forceinline__ v16bf ldsB_T(const unsigned short* p, int row0, int stride,
                                               int col0, int lane) {
    const unsigned short* r = p + (row0 + (lane & 15)) * stride + col0 + (lane >> 4) * 8;
    BF16x16 t;
    t.q[0] = *(const uint4*)(r);
    t.q[1] = *(const uint4*)(r + 16);
    return t.v;
}

// B-fragment from pre-converted row-major bf16 weights in global memory (2x b128, zero VALU)
static __device__ __forceinline__ v16bf gblB16(const unsigned short* __restrict__ W,
                                               int row0, int col0, int lane) {
    const unsigned short* r = W + (size_t)(row0 + (lane & 15)) * HH + col0 + (lane >> 4) * 8;
    BF16x16 t;
    t.q[0] = *(const uint4*)(r);
    t.q[1] = *(const uint4*)(r + 16);
    return t.v;
}

// pack 8 consecutive f32 accumulator elements -> 8 bf16 -> one b128 store
static __device__ __forceinline__ void store8bf(unsigned short* dst, const v8f acc, float mul) {
    union { unsigned short u[8]; uint4 q; } t;
#pragma unroll
    for (int v = 0; v < 8; ++v) t.u[v] = f2bf(acc[v] * mul);
    *(uint4*)dst = t.q;
}

// ---- prep: bulk f32 -> bf16 conversion (bandwidth bound, runs once) ----
__global__ void f32_to_bf16_kernel(const float* __restrict__ src,
                                   unsigned short* __restrict__ dst, long n) {
    long i = ((long)blockIdx.x * blockDim.x + threadIdx.x) * 4;
    const long stride = (long)gridDim.x * blockDim.x * 4;
    for (; i < n; i += stride) {
        float4 f = *(const float4*)(src + i);
        union { unsigned short u[4]; uint2 q; } t;
        t.u[0] = f2bf(f.x); t.u[1] = f2bf(f.y); t.u[2] = f2bf(f.z); t.u[3] = f2bf(f.w);
        *(uint2*)(dst + i) = t.q;
    }
}

__global__ void zero_ws_kernel(float* ws, long n) {
    long i = (long)blockIdx.x * blockDim.x + threadIdx.x;
    const long stride = (long)gridDim.x * blockDim.x;
    for (; i < n; i += stride) ws[i] = 0.0f;
}

__global__ void cnt_kernel(const int* __restrict__ pidx, float* __restrict__ cnt) {
    const int p = blockIdx.x;
    const int t = threadIdx.x;
    if (t < CORE_ROWS) atomicAdd(cnt + pidx[p * SS + t], 1.0f);
}

__global__ __launch_bounds__(256) void attn_kernel(
    const unsigned short* __restrict__ xbf, const int* __restrict__ pidx,
    const unsigned short* __restrict__ Wq, const float* __restrict__ bq,
    const unsigned short* __restrict__ Wk, const float* __restrict__ bk,
    const unsigned short* __restrict__ Wv, const float* __restrict__ bv,
    const unsigned short* __restrict__ Wo, const float* __restrict__ bo,
    float* __restrict__ accbuf)
{
    extern __shared__ unsigned char smem[];
    unsigned short* sPX  = (unsigned short*)smem;     // 256x264 bf16 : 132 KB (padded)
    unsigned short* sAtt = sPX  + SS * PX_STR;        // 256x264 bf16 : 132 KB (padded)
    unsigned short* sQ   = sAtt + SS * AT_STR;        // 256x32  bf16 :  16 KB
    unsigned short* sK   = sQ   + SS * DD;            // 256x32  bf16 :  16 KB
    unsigned short* sVT  = sK   + SS * DD;            // 32x264  bf16 : 16.5 KB (V^T, padded)
    int*            sIdx = (int*)(sVT + DD * VT_STRIDE);

    const int p    = blockIdx.x;
    const int tid  = threadIdx.x;
    const int lane = tid & 31;
    const int wave = tid >> 5;
    const int hf   = lane >> 4;

    // ---- Phase A: async gather of bf16 rows straight into LDS (ASYNCcnt path) ----
    for (int i = wave; i < SS; i += 8) {
        const int g = pidx[p * SS + i];
        if (lane == 0) sIdx[i] = g;
        const unsigned short* src = xbf + (size_t)g * HH + lane * 8;   // 16 B per lane
        unsigned ldsoff = (unsigned)(size_t)(sPX + i * PX_STR + lane * 8);
        unsigned long long ga = (unsigned long long)(size_t)src;
        asm volatile("global_load_async_to_lds_b128 %0, %1, off"
                     :: "v"(ldsoff), "v"(ga) : "memory");
    }
    asm volatile("s_wait_asynccnt 0x0" ::: "memory");
    __syncthreads();

    const float scale = 0.17677669529663687f; // 1/sqrt(D), folded into Q

    for (int h = 0; h < 8; ++h) {
        // ---- Phase B1: Q/K/V head projections; nt-pairs share one A-fragment ----
        // 48 tile-pairs: {Q,K,V} x (16 M-tiles), each pair = N-tiles {0,1}
        for (int t = wave; t < 48; t += 8) {
            const int mat = t / 16, mt = t % 16;
            const unsigned short* W = (mat == 0) ? Wq : (mat == 1) ? Wk : Wv;
            const float* bia        = (mat == 0) ? bq : (mat == 1) ? bk : bv;
            const float bv0 = bia[h * DD +      (lane & 15)];
            const float bv1 = bia[h * DD + 16 + (lane & 15)];
            v8f acc0, acc1;
#pragma unroll
            for (int v = 0; v < 8; ++v) { acc0[v] = bv0; acc1[v] = bv1; }
#pragma unroll
            for (int kk = 0; kk < HH; kk += 32) {
                v16bf a  = ldsA(sPX, mt * 16, PX_STR, kk, lane);
                v16bf b0 = gblB16(W, h * DD,      kk, lane);
                v16bf b1 = gblB16(W, h * DD + 16, kk, lane);
                acc0 = wmma_bf16(a, b0, acc0);
                acc1 = wmma_bf16(a, b1, acc1);
            }
            if (mat == 2) {
                // V stored transposed, 8 consecutive rows per lane -> packed b128 stores
                const int d0 = (lane & 15), d1 = 16 + (lane & 15);
                store8bf(sVT + d0 * VT_STRIDE + mt * 16 + hf * 8, acc0, 1.0f);
                store8bf(sVT + d1 * VT_STRIDE + mt * 16 + hf * 8, acc1, 1.0f);
            } else {
                unsigned short* dst = (mat == 0) ? sQ : sK;
                const float m = (mat == 0) ? scale : 1.0f;   // fold 1/sqrt(D) into Q
#pragma unroll
                for (int v = 0; v < 8; ++v) {
                    const int row = mt * 16 + v + hf * 8;
                    dst[row * DD +      (lane & 15)] = f2bf(acc0[v] * m);
                    dst[row * DD + 16 + (lane & 15)] = f2bf(acc1[v] * m);
                }
            }
        }
        __syncthreads();

        // ---- Phase B2: attention (scores transposed: scT[j,m] = K_j . Q_m) ----
        for (int rt = wave; rt < 16; rt += 8) {
            v8f scT[16];
            v16bf bQ = ldsB_T(sQ, rt * 16, DD, 0, lane); // B[d][m] = Q[rt*16+m][d] (pre-scaled)
#pragma unroll
            for (int ct = 0; ct < 16; ++ct) {
                v16bf aK = ldsA(sK, ct * 16, DD, 0, lane);
                v8f z;
#pragma unroll
                for (int v = 0; v < 8; ++v) z[v] = 0.0f;
                scT[ct] = wmma_bf16(aK, bQ, z);
            }
            // softmax over j for each query m (in-lane + one xor-16 combine)
            float rmax = -3.0e38f;
#pragma unroll
            for (int ct = 0; ct < 16; ++ct)
#pragma unroll
                for (int v = 0; v < 8; ++v) rmax = fmaxf(rmax, scT[ct][v]);
            rmax = fmaxf(rmax, __shfl_xor(rmax, 16, 32));
            float rsum = 0.0f;
#pragma unroll
            for (int ct = 0; ct < 16; ++ct)
#pragma unroll
                for (int v = 0; v < 8; ++v) {
                    float e = __expf(scT[ct][v] - rmax);
                    scT[ct][v] = e;
                    rsum += e;
                }
            rsum += __shfl_xor(rsum, 16, 32);
            const float inv = 1.0f / rsum;
#pragma unroll
            for (int ct = 0; ct < 16; ++ct)
#pragma unroll
                for (int v = 0; v < 8; ++v) scT[ct][v] *= inv;

            // att^T = V^T(16x32) x attnT(32x16); one attnT fragment feeds both d-tiles
            v8f oacc0, oacc1;
#pragma unroll
            for (int v = 0; v < 8; ++v) { oacc0[v] = 0.0f; oacc1[v] = 0.0f; }
#pragma unroll
            for (int c = 0; c < 8; ++c) {
                // build attnT B-fragment once per k-chunk, directly from score registers
                BF16x16 tb;
#pragma unroll
                for (int e = 0; e < 16; ++e) {
                    const int v  = e & 7;
                    const int hs = e >> 3;
                    float x0 = scT[2 * c + 0][v];
                    float x1 = scT[2 * c + 1][v];
                    float xs = hf ? x1 : x0;
                    float xo = __shfl_xor(xs, 16, 32);
                    tb.u[e] = f2bf(hs ? x1 : x0);
                }
                v16bf aV0 = ldsA(sVT, 0,  VT_STRIDE, c * 32, lane);
                v16bf aV1 = ldsA(sVT, 16, VT_STRIDE, c * 32, lane);
                oacc0 = wmma_bf16(aV0, tb.v, oacc0);
                oacc1 = wmma_bf16(aV1, tb.v, oacc1);
            }
            // store att^T tiles: 8 consecutive d per lane -> packed b128 stores
            {
                const int m = lane & 15;
                unsigned short* r = sAtt + (rt * 16 + m) * AT_STR + h * DD + hf * 8;
                store8bf(r,      oacc0, 1.0f);   // d = hf*8 + v        (tile 0)
                store8bf(r + 16, oacc1, 1.0f);   // d = 16 + hf*8 + v   (tile 1)
            }
        }
        __syncthreads();
    }

    // ---- Phase C: out = att @ Wo^T + bo for core rows (0..191); nt-pairs share A ----
    for (int t = wave; t < 96; t += 8) {
        const int mt = t / 8, np = t % 8;           // mt in 0..11 -> rows < 192 (core)
        v8f acc0, acc1;
        const float bv0 = bo[np * 32 +      (lane & 15)];
        const float bv1 = bo[np * 32 + 16 + (lane & 15)];
#pragma unroll
        for (int v = 0; v < 8; ++v) { acc0[v] = bv0; acc1[v] = bv1; }
#pragma unroll
        for (int kk = 0; kk < HH; kk += 32) {
            v16bf a  = ldsA(sAtt, mt * 16, AT_STR, kk, lane);
            v16bf b0 = gblB16(Wo, np * 32,      kk, lane);
            v16bf b1 = gblB16(Wo, np * 32 + 16, kk, lane);
            acc0 = wmma_bf16(a, b0, acc0);
            acc1 = wmma_bf16(a, b1, acc1);
        }
        const int col0 = np * 32 + (lane & 15);
#pragma unroll
        for (int v = 0; v < 8; ++v) {
            const int s = mt * 16 + v + hf * 8;   // < 192 == CORE rows only
            const int g = sIdx[s];
            atomicAdd(accbuf + (size_t)g * HH + col0,      acc0[v]);
            atomicAdd(accbuf + (size_t)g * HH + col0 + 16, acc1[v]);
        }
    }
}

__global__ __launch_bounds__(256) void finalize_kernel(
    const float* __restrict__ x, const float* __restrict__ accbuf,
    const float* __restrict__ cnt, const float* __restrict__ lg,
    const float* __restrict__ lb, float* __restrict__ out)
{
    const int row  = blockIdx.x * 8 + (threadIdx.x >> 5);
    const int lane = threadIdx.x & 31;
    const float* xr = x      + (size_t)row * HH + lane * 8;
    const float* ar = accbuf + (size_t)row * HH + lane * 8;
    const float is = 1.0f / fmaxf(cnt[row], 1.0f);

    float4 x0 = *(const float4*)(xr);
    float4 x1 = *(const float4*)(xr + 4);
    float4 a0 = *(const float4*)(ar);
    float4 a1 = *(const float4*)(ar + 4);
    float y[8];
    y[0] = x0.x + a0.x * is; y[1] = x0.y + a0.y * is;
    y[2] = x0.z + a0.z * is; y[3] = x0.w + a0.w * is;
    y[4] = x1.x + a1.x * is; y[5] = x1.y + a1.y * is;
    y[6] = x1.z + a1.z * is; y[7] = x1.w + a1.w * is;

    float s = 0.0f, s2 = 0.0f;
#pragma unroll
    for (int k = 0; k < 8; ++k) { s += y[k]; s2 += y[k] * y[k]; }
#pragma unroll
    for (int m = 1; m < 32; m <<= 1) {
        s  += __shfl_xor(s,  m, 32);
        s2 += __shfl_xor(s2, m, 32);
    }
    const float mu  = s * (1.0f / 256.0f);
    const float var = s2 * (1.0f / 256.0f) - mu * mu;
    const float r   = rsqrtf(var + 1e-5f);
    const int col = lane * 8;
#pragma unroll
    for (int k = 0; k < 8; ++k)
        out[(size_t)row * HH + col + k] = (y[k] - mu) * r * lg[col + k] + lb[col + k];
}

extern "C" void kernel_launch(void* const* d_in, const int* in_sizes, int n_in,
                              void* d_out, int out_size, void* d_ws, size_t ws_size,
                              hipStream_t stream) {
    const float* x   = (const float*)d_in[0];
    const int*  pidx = (const int*)d_in[1];
    // d_in[2] = core_mask: by construction core_mask[p,s] == (s < CORE); handled statically.
    const float* Wq = (const float*)d_in[3];  const float* bq = (const float*)d_in[4];
    const float* Wk = (const float*)d_in[5];  const float* bk = (const float*)d_in[6];
    const float* Wv = (const float*)d_in[7];  const float* bv = (const float*)d_in[8];
    const float* Wo = (const float*)d_in[9];  const float* bo = (const float*)d_in[10];
    const float* lg = (const float*)d_in[11]; const float* lb = (const float*)d_in[12];
    float* out = (float*)d_out;

    // ws layout: [accbuf f32 N*H][cnt f32 N][xbf bf16 N*H][wq|wk|wv|wo bf16 H*H each]
    float* accbuf = (float*)d_ws;
    float* cnt    = accbuf + (size_t)NTOK * HH;
    unsigned short* xbf = (unsigned short*)(cnt + NTOK);
    unsigned short* wqb = xbf + (size_t)NTOK * HH;
    unsigned short* wkb = wqb + (size_t)HH * HH;
    unsigned short* wvb = wkb + (size_t)HH * HH;
    unsigned short* wob = wvb + (size_t)HH * HH;

    const long ztotal = (long)NTOK * HH + NTOK;
    const long xn = (long)NTOK * HH;
    const long wn = (long)HH * HH;

    const int ldsBytes = (SS * PX_STR + SS * AT_STR + 2 * SS * DD + DD * VT_STRIDE) * 2
                         + SS * 4; // 321024 B <= 320 KB WGP LDS
    (void)hipFuncSetAttribute((const void*)attn_kernel,
                              hipFuncAttributeMaxDynamicSharedMemorySize, ldsBytes);

    f32_to_bf16_kernel<<<4096, 256, 0, stream>>>(x,  xbf, xn);
    f32_to_bf16_kernel<<<64,   256, 0, stream>>>(Wq, wqb, wn);
    f32_to_bf16_kernel<<<64,   256, 0, stream>>>(Wk, wkb, wn);
    f32_to_bf16_kernel<<<64,   256, 0, stream>>>(Wv, wvb, wn);
    f32_to_bf16_kernel<<<64,   256, 0, stream>>>(Wo, wob, wn);
    zero_ws_kernel<<<2048, 256, 0, stream>>>(accbuf, ztotal);
    cnt_kernel<<<PP, 256, 0, stream>>>(pidx, cnt);
    attn_kernel<<<PP, 256, ldsBytes, stream>>>(xbf, pidx, wqb, bq, wkb, bk, wvb, bv, wob, bo, accbuf);
    finalize_kernel<<<NTOK / 8, 256, 0, stream>>>(x, accbuf, cnt, lg, lb, out);
}
